// SimpleRNN_12472585027616
// MI455X (gfx1250) — matrix operand order repaired and checked by reading, not compiled, not run on hardware
//
#include <hip/hip_runtime.h>
#include <math.h>
#include <stdint.h>

// Problem constants (from reference): T=2048, B=64, N_IN=128, H=256, N_OUT=128
#define T_STEPS 2048
#define BATCH   64
#define NIN     128
#define HDIM    256
#define NOUT    128

typedef __attribute__((ext_vector_type(16))) __bf16 v16bf;
typedef __attribute__((ext_vector_type(8)))  float  v8f;

// ---------------------------------------------------------------------------
// Hardware tanh: CDNA5 has V_TANH_F32 in the TRANS group. Prefer the builtin
// (compiler handles TRANS hazards); fall back to inline asm with a v_nop to
// cover the "1 independent op after TRANS before result use" rule.
// ---------------------------------------------------------------------------
static __device__ __forceinline__ float fast_tanh(float x) {
#if __has_builtin(__builtin_amdgcn_tanhf)
  return __builtin_amdgcn_tanhf(x);
#else
  float y;
  asm("v_tanh_f32 %0, %1\n\tv_nop" : "=v"(y) : "v"(x));
  return y;
#endif
}

// ---------------------------------------------------------------------------
// Fragment conversion helpers (f32 global -> bf16 WMMA fragments, wave32)
//
// A-matrix 16x32 bf16 layout: lane L holds row m = L%16; element j maps to
//   k = 32*kc + (j/8)*16 + 8*(L/16) + (j%8)   -> two contiguous 8-float runs.
// B-matrix 32x16 bf16 layout: lane L holds col n = L%16; element j maps to
//   k = 32*kc + (j/8)*16 + 8*(L/16) + (j%8)   -> two contiguous 8-float runs.
// ---------------------------------------------------------------------------
static __device__ __forceinline__ v16bf cvt_contig16(const float* __restrict__ p) {
  v8f lo = *(const v8f*)p;
  v8f hi = *(const v8f*)(p + 16);
  v16bf r;
#pragma unroll
  for (int i = 0; i < 8; ++i) { r[i] = (__bf16)lo[i]; r[i + 8] = (__bf16)hi[i]; }
  return r;
}

static __device__ __forceinline__ v16bf cvt_a_frag(const float* __restrict__ row,
                                                   int kc, int lhi) {
  const float* p0 = row + 32 * kc + 8 * lhi;       // j = 0..7
  v8f lo = *(const v8f*)p0;
  v8f hi = *(const v8f*)(p0 + 16);                 // j = 8..15 (k += 16)
  v16bf r;
#pragma unroll
  for (int i = 0; i < 8; ++i) { r[i] = (__bf16)lo[i]; r[i + 8] = (__bf16)hi[i]; }
  return r;
}

// ---------------------------------------------------------------------------
// Kernel A: x_proj[t,b,:] = x[t,b,:] @ W_ih^T + (b_ih + b_hh)
// Grid-wide bf16 WMMA GEMM: M = T*B = 131072, K = 128, N = 256.
// Output stored tile-major in RAW C-fragment order so the scan kernel can
// load it straight into WMMA accumulators. Regular (RT) stores on purpose:
// x_proj (128 MB) fits the 192 MB L2, so the scan streams it from L2.
// ---------------------------------------------------------------------------
__global__ __launch_bounds__(256) void rnn_xproj_kernel(
    const float* __restrict__ x, const float* __restrict__ w_ih,
    const float* __restrict__ b_ih, const float* __restrict__ b_hh,
    float* __restrict__ xp) {
  const int tid  = threadIdx.x;
  const int wv   = tid >> 5;
  const int lane = tid & 31;
  const int lrow = lane & 15;
  const int lhi  = lane >> 4;

  const int gw  = blockIdx.x * 8 + wv;   // 0 .. 32767
  const int MT  = gw >> 2;               // 0 .. 8191  (16 rows of T*B each)
  const int ntb = gw & 3;                // 4 N-tiles per wave

  const float* xrow = x + (size_t)(16 * MT + lrow) * NIN;

  v8f acc[4];
#pragma unroll
  for (int q = 0; q < 4; ++q)
#pragma unroll
    for (int r = 0; r < 8; ++r) acc[q][r] = 0.0f;

#pragma unroll
  for (int kc = 0; kc < NIN / 32; ++kc) {
    v16bf a = cvt_a_frag(xrow, kc, lhi);
#pragma unroll
    for (int q = 0; q < 4; ++q) {
      const int nt = ntb * 4 + q;
      v16bf b = cvt_contig16(w_ih + (size_t)(16 * nt + lrow) * NIN + 32 * kc + 8 * lhi);
      acc[q] = __builtin_amdgcn_wmma_f32_16x16x32_bf16(
          false, a, false, b, (short)0, acc[q], false, false);
    }
  }

#pragma unroll
  for (int q = 0; q < 4; ++q) {
    const int nt   = ntb * 4 + q;
    const float bb = b_ih[16 * nt + lrow] + b_hh[16 * nt + lrow];  // n = lane%16
#pragma unroll
    for (int r = 0; r < 8; ++r) acc[q][r] += bb;
    *(v8f*)(xp + ((size_t)MT * 16 + nt) * 256 + lane * 8) = acc[q];
  }
}

// ---------------------------------------------------------------------------
// Kernel B: sequential scan, single persistent workgroup (16 waves / 1 WGP).
//   h_t = tanh(x_proj_t + h_{t-1} @ W_hh^T);  out_t = h_t @ W_out^T + b_out
// - W_hh / W_out live in registers as pre-converted bf16 B-fragments.
// - h double-buffered in LDS (2 x 32KB) in A-fragment layout (pure 32B loads).
// - Split barrier per step; next step's x_proj accumulator loads are issued
//   between s_barrier_signal and s_barrier_wait to hide global latency.
// ---------------------------------------------------------------------------
__global__ __launch_bounds__(512) void rnn_scan_kernel(
    const float* __restrict__ xp, const float* __restrict__ w_hh,
    const float* __restrict__ w_out, const float* __restrict__ b_out,
    float* __restrict__ out) {
  __shared__ char hbuf[2 * 32768];   // [buf][mt(4)][kc(8)][lane(32)][32B]

  const int tid  = threadIdx.x;
  const int wv   = tid >> 5;   // 0..15 : recurrent N-tile (h columns 16wv..16wv+15)
  const int lane = tid & 31;
  const int lrow = lane & 15;
  const int lhi  = lane >> 4;

  // zero h_{-1} (buffer 0)
  for (int i = tid; i < 32768 / 4; i += 512) ((uint32_t*)hbuf)[i] = 0u;

  // Persistent bf16 B-fragments: W_hh (this wave's N-tile = wv), all 8 K-chunks.
  v16bf whh[8];
#pragma unroll
  for (int kc = 0; kc < 8; ++kc)
    whh[kc] = cvt_contig16(w_hh + (size_t)(16 * wv + lrow) * HDIM + 32 * kc + 8 * lhi);

  // Output projection assignment: 32 tiles (4 mt x 8 nto) over 16 waves.
  const int nto = wv & 7;
  const int mtb = (wv >> 3) << 1;   // this wave handles mt = mtb, mtb+1
  v16bf wob[8];
#pragma unroll
  for (int kc = 0; kc < 8; ++kc)
    wob[kc] = cvt_contig16(w_out + (size_t)(16 * nto + lrow) * HDIM + 32 * kc + 8 * lhi);
  const float obias = b_out[16 * nto + lrow];

  // D-fragment -> A-fragment scatter constants for h stores (consumer lane L',
  // element j reads exactly h[16mt + L'%16, 16wv + lrow]):
  const int kc_w    = wv >> 1;                       // K-chunk of this wave's h columns
  const int jst     = ((wv & 1) << 3) + (lrow & 7);  // j slot
  const int lanep_b = (lrow >= 8) ? 16 : 0;          // lane' high half

  __syncthreads();

  // Software pipeline: accumulators for step t are loaded during step t-1.
  v8f acc[4];
#pragma unroll
  for (int mt = 0; mt < 4; ++mt)
    acc[mt] = *(const v8f*)(xp + ((size_t)mt * 16 + wv) * 256 + lane * 8);

  for (int t = 0; t < T_STEPS; ++t) {
    char* rbuf = hbuf + ((t & 1) << 15);
    char* wbuf = hbuf + (((t & 1) ^ 1) << 15);

    // ---- recurrent GEMM: acc[mt] (= x_proj tile + bias) += h_{t-1} @ W_hh^T
#pragma unroll
    for (int mt = 0; mt < 4; ++mt) {
#pragma unroll
      for (int kc = 0; kc < 8; ++kc) {
        v16bf a = *(const v16bf*)(rbuf + ((mt * 8 + kc) << 10) + (lane << 5));
        acc[mt] = __builtin_amdgcn_wmma_f32_16x16x32_bf16(
            false, a, false, whh[kc], (short)0, acc[mt], false, false);
      }
    }

    // ---- tanh (hardware V_TANH_F32) + scatter h_t into the opposite buffer
#pragma unroll
    for (int mt = 0; mt < 4; ++mt) {
#pragma unroll
      for (int r = 0; r < 8; ++r) {
        const int lanep = ((lhi << 3) + r) + lanep_b;
        *((__bf16*)(wbuf + ((mt * 8 + kc_w) << 10) + (lanep << 5) + (jst << 1))) =
            (__bf16)fast_tanh(acc[mt][r]);
      }
    }

    // ---- split barrier: signal, overlap next step's global loads, then wait.
    // s_wait_dscnt 0 makes our ds_stores visible (and our rbuf ds_loads done)
    // before signaling -- same guarantee __syncthreads provides.
    asm volatile("s_wait_dscnt 0x0\n\ts_barrier_signal -1" ::: "memory");

    v8f nacc[4];
    if (t + 1 < T_STEPS) {
#pragma unroll
      for (int mt = 0; mt < 4; ++mt)
        nacc[mt] = *(const v8f*)(xp + (((size_t)(t + 1) * 4 + mt) * 16 + wv) * 256 + lane * 8);
      // prefetch one step further ahead into L0/L2 (global_prefetch_b8)
#pragma unroll
      for (int mt = 0; mt < 4; ++mt)
        __builtin_prefetch(xp + (((size_t)(t + 2) * 4 + mt) * 16 + wv) * 256 + lane * 8, 0, 1);
    } else {
#pragma unroll
      for (int mt = 0; mt < 4; ++mt)
#pragma unroll
        for (int r = 0; r < 8; ++r) nacc[mt][r] = 0.0f;
    }

    asm volatile("s_barrier_wait -1" ::: "memory");

    // ---- output projection: out_t = h_t @ W_out^T + b_out
#pragma unroll
    for (int i = 0; i < 2; ++i) {
      const int mt = mtb + i;
      v8f c;
#pragma unroll
      for (int r = 0; r < 8; ++r) c[r] = obias;
#pragma unroll
      for (int kc = 0; kc < 8; ++kc) {
        v16bf a = *(const v16bf*)(wbuf + ((mt * 8 + kc) << 10) + (lane << 5));
        c = __builtin_amdgcn_wmma_f32_16x16x32_bf16(
            false, a, false, wob[kc], (short)0, c, false, false);
      }
      // Non-temporal stores: d_out is write-once; keep L2 for x_proj.
#pragma unroll
      for (int r = 0; r < 8; ++r)
        __builtin_nontemporal_store(
            c[r],
            out + ((size_t)t * BATCH + 16 * mt + (lhi << 3) + r) * NOUT + 16 * nto + lrow);
    }

#pragma unroll
    for (int mt = 0; mt < 4; ++mt) acc[mt] = nacc[mt];
  }
}

// ---------------------------------------------------------------------------
// Launch: inputs in setup_inputs() order:
//   0:x [T,B,NIN] 1:w_ih [H,NIN] 2:w_hh [H,H] 3:b_ih [H] 4:b_hh [H]
//   5:w_out [NOUT,H] 6:b_out [NOUT]
// Workspace: x_proj in raw C-fragment order, T*B*H f32 = 128 MiB.
// ---------------------------------------------------------------------------
extern "C" void kernel_launch(void* const* d_in, const int* in_sizes, int n_in,
                              void* d_out, int out_size, void* d_ws, size_t ws_size,
                              hipStream_t stream) {
  (void)in_sizes; (void)n_in; (void)out_size; (void)ws_size;
  const float* x     = (const float*)d_in[0];
  const float* w_ih  = (const float*)d_in[1];
  const float* w_hh  = (const float*)d_in[2];
  const float* b_ih  = (const float*)d_in[3];
  const float* b_hh  = (const float*)d_in[4];
  const float* w_out = (const float*)d_in[5];
  const float* b_out = (const float*)d_in[6];
  float* out = (float*)d_out;
  float* xp  = (float*)d_ws;   // requires T*B*H*4 = 134,217,728 bytes of scratch

  // Phase 1: grid-wide input projection (32768 waves = 8192 M-tiles x 4)
  rnn_xproj_kernel<<<4096, 256, 0, stream>>>(x, w_ih, b_ih, b_hh, xp);
  // Phase 2+3: persistent single-WGP scan with fused output projection
  rnn_scan_kernel<<<1, 512, 0, stream>>>(xp, w_hh, w_out, b_out, out);
}
